// SwinTransformerBlock_22969485099058
// MI455X (gfx1250) — compile-verified, matrix operand order checked by reading, not run on hardware
//
#include <hip/hip_runtime.h>
#include <hip/hip_bf16.h>
#include <math.h>

typedef __bf16 bf16_t;
typedef __bf16 v16bf __attribute__((ext_vector_type(16)));
typedef __bf16 v8bf  __attribute__((ext_vector_type(8)));
typedef float  v8f   __attribute__((ext_vector_type(8)));

#define CC    192
#define POSN  32768     // total positions = B*H*W = windows*64
#define NWIN  512
#define NHEAD 6
#define HD    32
#define HID2  1536
#define HIDN  768
#define ATT_SCALE 0.17677669529663687f   // 32^-0.5

__device__ __forceinline__ float gelu_f(float x) {
  return 0.5f * x * (1.0f + erff(x * 0.70710678118654752f));
}

// concat two contiguous 8-elem runs into a WMMA 16-bit fragment
__device__ __forceinline__ v16bf load_frag16(const bf16_t* p0, const bf16_t* p1) {
  v8bf lo = *(const v8bf*)p0;
  v8bf hi = *(const v8bf*)p1;
  return __builtin_shufflevector(lo, hi, 0, 1, 2, 3, 4, 5, 6, 7,
                                 8, 9, 10, 11, 12, 13, 14, 15);
}

__constant__ float COS8[8] = {
  1.0f, 0.70710678f, 0.0f, -0.70710678f, -1.0f, -0.70710678f, 0.0f, 0.70710678f
};

// ---------------------------------------------------------------- weights
// Pack W (O=N rows, K cols, f32) into WMMA-B fragment order:
// F[((kt*(N/16)+nt)*32 + lane)*16 + e], lane: n=lane&15, kb=(lane>>4)*8,
// e<8 -> k=kb+e ; e>=8 -> k=16+kb+(e-8).  One 32B load per lane per fragment.
__global__ void convert_frag_kernel(const float* __restrict__ W, bf16_t* __restrict__ F,
                                    int N, int K) {
  size_t idx = (size_t)blockIdx.x * blockDim.x + threadIdx.x;
  if (idx >= (size_t)N * K) return;
  int e    = (int)(idx & 15);
  int lane = (int)((idx >> 4) & 31);
  size_t t = idx >> 9;                 // kt*(N/16) + nt
  int ntiles = N >> 4;
  int kt = (int)(t / ntiles), nt = (int)(t % ntiles);
  int nl = lane & 15, kb = (lane >> 4) << 3;
  int kl = (e < 8) ? (kb + e) : (16 + kb + (e - 8));
  F[idx] = (bf16_t)W[(size_t)(nt * 16 + nl) * K + kt * 32 + kl];
}

// bias[h][i][j] = rpb[relidx(i,j)*HEADS + h]
__global__ void bias_table_kernel(const float* __restrict__ rpb, float* __restrict__ tab) {
  int idx = blockIdx.x * blockDim.x + threadIdx.x;  // h*4096 + i*64 + j
  if (idx >= NHEAD * 64 * 64) return;
  int j = idx & 63, i = (idx >> 6) & 63, h = idx >> 12;
  int yi = i >> 3, xi = i & 7, yj = j >> 3, xj = j & 7;
  int ridx = (yi - yj + 7) * 15 + (xi - xj + 7);
  tab[idx] = rpb[ridx * NHEAD + h];
}

// ---------------------------------------------------------------- layernorms
__global__ void ln1_window_kernel(const float* __restrict__ x,
                                  const float* __restrict__ g, const float* __restrict__ be,
                                  float* __restrict__ xw_f, bf16_t* __restrict__ xw_b) {
  int wave = threadIdx.x >> 5, lane = threadIdx.x & 31;
  int t = blockIdx.x * (blockDim.x >> 5) + wave;
  if (t >= POSN) return;
  const float* row = x + (size_t)t * CC;
  float v[6], s = 0.f;
#pragma unroll
  for (int i = 0; i < 6; ++i) { v[i] = row[lane + 32 * i]; s += v[i]; }
#pragma unroll
  for (int off = 16; off > 0; off >>= 1) s += __shfl_xor(s, off, 32);
  float mean = s * (1.0f / CC);
  float vs = 0.f;
#pragma unroll
  for (int i = 0; i < 6; ++i) { float d = v[i] - mean; vs += d * d; }
#pragma unroll
  for (int off = 16; off > 0; off >>= 1) vs += __shfl_xor(vs, off, 32);
  float rstd = rsqrtf(vs * (1.0f / CC) + 1e-5f);
  int b  = t >> 14, l = t & 16383;
  int hh = l >> 7, ww = l & 127;
  int wpos = ((b * 256 + (hh >> 3) * 16 + (ww >> 3)) << 6) + ((hh & 7) << 3) + (ww & 7);
  float* of = xw_f + (size_t)wpos * CC;
  bf16_t* ob = xw_b + (size_t)wpos * CC;
#pragma unroll
  for (int i = 0; i < 6; ++i) {
    int c = lane + 32 * i;
    float y = (v[i] - mean) * rstd * g[c] + be[c];
    of[c] = y; ob[c] = (bf16_t)y;
  }
}

__global__ void ln2_kernel(const float* __restrict__ x,
                           const float* __restrict__ g, const float* __restrict__ be,
                           bf16_t* __restrict__ out) {
  int wave = threadIdx.x >> 5, lane = threadIdx.x & 31;
  int t = blockIdx.x * (blockDim.x >> 5) + wave;
  if (t >= POSN) return;
  const float* row = x + (size_t)t * CC;
  float v[6], s = 0.f;
#pragma unroll
  for (int i = 0; i < 6; ++i) { v[i] = row[lane + 32 * i]; s += v[i]; }
#pragma unroll
  for (int off = 16; off > 0; off >>= 1) s += __shfl_xor(s, off, 32);
  float mean = s * (1.0f / CC);
  float vs = 0.f;
#pragma unroll
  for (int i = 0; i < 6; ++i) { float d = v[i] - mean; vs += d * d; }
#pragma unroll
  for (int off = 16; off > 0; off >>= 1) vs += __shfl_xor(vs, off, 32);
  float rstd = rsqrtf(vs * (1.0f / CC) + 1e-5f);
  bf16_t* ob = out + (size_t)t * CC;
#pragma unroll
  for (int i = 0; i < 6; ++i) {
    int c = lane + 32 * i;
    ob[c] = (bf16_t)((v[i] - mean) * rstd * g[c] + be[c]);
  }
}

// ---------------------------------------------------------------- real 8x8 DFT
__global__ void fft_fwd_kernel(const float* __restrict__ xw, bf16_t* __restrict__ out) {
  __shared__ float tile[4][64];
  int local = threadIdx.x & 63, tl = threadIdx.x >> 6;
  size_t tileIdx = (size_t)blockIdx.x * 4 + tl;     // win*192 + ch
  int win = (int)(tileIdx / CC), ch = (int)(tileIdx % CC);
  tile[tl][local] = xw[((size_t)win * 64 + local) * CC + ch];
  __syncthreads();
  int u = local >> 3, v = local & 7;
  float acc = 0.f;
#pragma unroll
  for (int m = 0; m < 8; ++m)
#pragma unroll
    for (int n = 0; n < 8; ++n)
      acc += tile[tl][m * 8 + n] * COS8[(u * m + v * n) & 7];
  out[((size_t)win * 64 + local) * CC + ch] = (bf16_t)acc;
}

__global__ void fft_inv_kernel(const float* __restrict__ xin, float* __restrict__ out) {
  __shared__ float tile[4][64];
  int local = threadIdx.x & 63, tl = threadIdx.x >> 6;
  size_t tileIdx = (size_t)blockIdx.x * 4 + tl;
  int win = (int)(tileIdx / CC), ch = (int)(tileIdx % CC);
  tile[tl][local] = xin[((size_t)win * 64 + local) * CC + ch];
  __syncthreads();
  int u = local >> 3, v = local & 7;
  float acc = 0.f;
#pragma unroll
  for (int m = 0; m < 8; ++m)
#pragma unroll
    for (int n = 0; n < 8; ++n)
      acc += tile[tl][m * 8 + n] * COS8[(u * m + v * n) & 7];
  out[(size_t)win * (CC * 64) + (size_t)ch * 64 + local] = acc * (1.0f / 64.0f);
}

// ---------------------------------------------------------------- bf16 WMMA GEMM
// C[M,N] = A[M,K](bf16) * Bfrag[K,N](fragment-packed bf16) (+bias)(+gelu).
// Block = 8 waves; wave computes 16 rows x 64 cols (4 WMMA N-tiles, 4x A reuse).
__global__ void gemm_bf16_kernel(const bf16_t* __restrict__ A,
                                 const bf16_t* __restrict__ Bfrag,
                                 const float* __restrict__ bias,
                                 float* __restrict__ Cf,
                                 bf16_t* __restrict__ Cb,
                                 int M, int N, int K, int act) {
  int lane = threadIdx.x & 31;
  int wave = threadIdx.x >> 5;
  int col0 = blockIdx.x * 64;
  int row0 = (blockIdx.y * 8 + wave) * 16;

  int m  = lane & 15;
  int kb = (lane >> 4) << 3;
  int ntiles = N >> 4;
  int nt0 = col0 >> 4;

  v8f acc0 = {}, acc1 = {}, acc2 = {}, acc3 = {};
  for (int k0 = 0; k0 < K; k0 += 32) {
    const bf16_t* arow = A + (size_t)(row0 + m) * K + k0;
    __builtin_prefetch((const void*)(arow + 64), 0, 1);
    v16bf a = load_frag16(arow + kb, arow + 16 + kb);
    const bf16_t* fb = Bfrag + ((((size_t)(k0 >> 5)) * ntiles + nt0) << 9) + (lane << 4);
    v16bf b0 = *(const v16bf*)(fb);
    v16bf b1 = *(const v16bf*)(fb + 512);
    v16bf b2 = *(const v16bf*)(fb + 1024);
    v16bf b3 = *(const v16bf*)(fb + 1536);
    acc0 = __builtin_amdgcn_wmma_f32_16x16x32_bf16(false, a, false, b0, (short)0, acc0, false, false);
    acc1 = __builtin_amdgcn_wmma_f32_16x16x32_bf16(false, a, false, b1, (short)0, acc1, false, false);
    acc2 = __builtin_amdgcn_wmma_f32_16x16x32_bf16(false, a, false, b2, (short)0, acc2, false, false);
    acc3 = __builtin_amdgcn_wmma_f32_16x16x32_bf16(false, a, false, b3, (short)0, acc3, false, false);
  }
  int n  = lane & 15;
  int mo = (lane >> 4) << 3;
  v8f accs[4] = {acc0, acc1, acc2, acc3};
#pragma unroll
  for (int nt = 0; nt < 4; ++nt) {
    int cidx = col0 + nt * 16 + n;
    float bv = bias ? bias[cidx] : 0.f;
#pragma unroll
    for (int r = 0; r < 8; ++r) {
      float v = accs[nt][r] + bv;
      if (act == 1) v = gelu_f(v);
      size_t idx = (size_t)(row0 + mo + r) * N + cidx;
      if (Cf) Cf[idx] = v;
      if (Cb) Cb[idx] = (bf16_t)v;
    }
  }
}

// ---------------------------------------------------------------- qkv depthwise 3x3 over (win, n) grid
__global__ void qkv_dw_kernel(const float* __restrict__ t, const float* __restrict__ w,
                              const float* __restrict__ bias, float* __restrict__ out) {
  size_t idx = (size_t)blockIdx.x * blockDim.x + threadIdx.x;  // pos*576 + ch
  if (idx >= (size_t)POSN * 576) return;
  int ch = (int)(idx % 576);
  size_t pos = idx / 576;
  int bb = (int)(pos >> 6), nn = (int)(pos & 63);
  float acc = bias[ch];
#pragma unroll
  for (int dy = -1; dy <= 1; ++dy) {
    int by = bb + dy; if ((unsigned)by >= (unsigned)NWIN) continue;
#pragma unroll
    for (int dx = -1; dx <= 1; ++dx) {
      int nx = nn + dx; if ((unsigned)nx >= 64u) continue;
      acc += w[ch * 9 + (dy + 1) * 3 + (dx + 1)] * t[((size_t)by * 64 + nx) * 576 + ch];
    }
  }
  out[idx] = acc;
}

// ---------------------------------------------------------------- q/k/v assembly; V stored transposed (d-major)
__global__ void qkv_assemble_kernel(const float* __restrict__ dwout,
                                    const float* __restrict__ xifft,
                                    bf16_t* __restrict__ q, bf16_t* __restrict__ k,
                                    bf16_t* __restrict__ vT) {
  size_t idx = (size_t)blockIdx.x * blockDim.x + threadIdx.x;  // ((win*6+h)*64+n)*32+d
  if (idx >= (size_t)NWIN * NHEAD * 64 * HD) return;
  int dd = (int)(idx & 31);
  size_t r1 = idx >> 5;
  int n = (int)(r1 & 63);
  size_t r2 = r1 >> 6;
  int h = (int)(r2 % NHEAD);
  int win = (int)(r2 / NHEAD);
  int flat = h * 2048 + n * 32 + dd;           // torch raw reshape of (C,8,8) -> (6,64,32)
  int c = flat >> 6, p = flat & 63;
  float xi = xifft[(size_t)win * (CC * 64) + (size_t)c * 64 + p];
  size_t row = (size_t)(win * 64 + n) * 576;
  int col = h * HD + dd;
  q[idx] = (bf16_t)((dwout[row + col] + xi) * ATT_SCALE);
  k[idx] = (bf16_t)(dwout[row + CC + col] + xi);
  vT[r2 * 2048 + (size_t)dd * 64 + n] = (bf16_t)(dwout[row + 2 * CC + col] + xi);
}

// ---------------------------------------------------------------- attention: one wave per (window, head)
__global__ void attention_kernel(const bf16_t* __restrict__ Qg,
                                 const bf16_t* __restrict__ Kg,
                                 const bf16_t* __restrict__ VTg,
                                 const float* __restrict__ biasTab,
                                 bf16_t* __restrict__ Og) {
  __shared__ float  S[64][64];
  __shared__ bf16_t P[64][64];
  int lane = threadIdx.x;
  int bh = blockIdx.x;
  int win = bh / NHEAD, h = bh % NHEAD;
  const bf16_t* Qp = Qg + (size_t)bh * 2048;
  const bf16_t* Kp = Kg + (size_t)bh * 2048;
  const bf16_t* Vp = VTg + (size_t)bh * 2048;   // [d][m], d-major
  const float* Bt = biasTab + (size_t)h * 4096;

  int mrow = lane & 15;
  int kb   = (lane >> 4) << 3;
  int mo   = (lane >> 4) << 3;
  int ncol = lane & 15;

  // S = Q(64x32) . K^T  -> 4x4 tiles, single K=32 wmma each
  for (int mt = 0; mt < 4; ++mt) {
    const bf16_t* qr = Qp + (mt * 16 + mrow) * 32;
    v16bf a = load_frag16(qr + kb, qr + 16 + kb);
    for (int nt = 0; nt < 4; ++nt) {
      const bf16_t* kr = Kp + (nt * 16 + ncol) * 32;      // B[k][n] = K[n][k]
      v16bf bfr = load_frag16(kr + kb, kr + 16 + kb);
      v8f c = {};
      c = __builtin_amdgcn_wmma_f32_16x16x32_bf16(false, a, false, bfr,
                                                  (short)0, c, false, false);
#pragma unroll
      for (int r = 0; r < 8; ++r) {
        int row = mt * 16 + mo + r, col = nt * 16 + ncol;
        S[row][col] = c[r] + Bt[row * 64 + col];
      }
    }
  }
  __syncthreads();
  // softmax rows: each lane handles rows lane and lane+32
#pragma unroll
  for (int rr = 0; rr < 2; ++rr) {
    int row = lane + rr * 32;
    float mx = -1e30f;
    for (int j = 0; j < 64; ++j) mx = fmaxf(mx, S[row][j]);
    float sum = 0.f;
    for (int j = 0; j < 64; ++j) sum += __expf(S[row][j] - mx);
    float inv = 1.0f / sum;
    for (int j = 0; j < 64; ++j) P[row][j] = (bf16_t)(__expf(S[row][j] - mx) * inv);
  }
  __syncthreads();
  // O = P(64x64) . V(64x32) -> 4x2 tiles, two chained K=32 wmmas
  for (int mt = 0; mt < 4; ++mt) {
    for (int nt = 0; nt < 2; ++nt) {
      v8f acc = {};
      for (int kt = 0; kt < 2; ++kt) {
        const bf16_t* pr = &P[mt * 16 + mrow][kt * 32];
        v16bf a = load_frag16(pr + kb, pr + 16 + kb);
        const bf16_t* vr = Vp + (nt * 16 + ncol) * 64 + kt * 32;  // B[k][n] = vT[n][k]
        v16bf bfr = load_frag16(vr + kb, vr + 16 + kb);
        acc = __builtin_amdgcn_wmma_f32_16x16x32_bf16(false, a, false, bfr,
                                                      (short)0, acc, false, false);
      }
#pragma unroll
      for (int r = 0; r < 8; ++r) {
        size_t row = (size_t)win * 64 + mt * 16 + mo + r;
        Og[row * CC + h * HD + nt * 16 + ncol] = (bf16_t)acc[r];
      }
    }
  }
}

// ---------------------------------------------------------------- residuals
__global__ void residual1_kernel(const float* __restrict__ x, const float* __restrict__ projw,
                                 float* __restrict__ x2) {
  size_t idx = (size_t)blockIdx.x * blockDim.x + threadIdx.x;
  if (idx >= (size_t)POSN * CC) return;
  int c = (int)(idx % CC);
  size_t t = idx / CC;
  int b = (int)(t >> 14), l = (int)(t & 16383);
  int hh = l >> 7, ww = l & 127;
  int wpos = ((b * 256 + (hh >> 3) * 16 + (ww >> 3)) << 6) + ((hh & 7) << 3) + (ww & 7);
  x2[idx] = x[idx] + projw[(size_t)wpos * CC + c];
}

// out = x2 + reshape_raw(ffn_out (B,C,H,W) -> (B,L,C))
__global__ void residual2_kernel(const float* __restrict__ x2, const float* __restrict__ ffn,
                                 float* __restrict__ out) {
  size_t idx = (size_t)blockIdx.x * blockDim.x + threadIdx.x;
  if (idx >= (size_t)POSN * CC) return;
  size_t per = (size_t)16384 * CC;
  int b = (int)(idx / per);
  size_t r = idx % per;              // = l*C + c = ch*16384 + p (raw flatten)
  int ch = (int)(r >> 14);
  int p  = (int)(r & 16383);
  out[idx] = x2[idx] + ffn[((size_t)(b << 14) + p) * CC + ch];
}

// ---------------------------------------------------------------- FFN depthwise stage 1
__global__ void ffn_dw1_kernel(const float* __restrict__ xin,
                               const float* __restrict__ w3, const float* __restrict__ w5,
                               float* __restrict__ x1o, float* __restrict__ x2o) {
  size_t idx = (size_t)blockIdx.x * blockDim.x + threadIdx.x;  // pos*1536 + ch
  if (idx >= (size_t)POSN * HID2) return;
  int ch = (int)(idx % HID2);
  size_t pos = idx / HID2;
  int p = (int)(pos & 16383);
  size_t bimg = pos >> 14;
  int hh = p >> 7, ww = p & 127;
  const float* base = xin + (bimg << 14) * HID2;
  float r3 = 0.f, r5 = 0.f;
#pragma unroll
  for (int dy = -1; dy <= 1; ++dy) {
    int y = hh + dy; if ((unsigned)y >= 128u) continue;
#pragma unroll
    for (int dx = -1; dx <= 1; ++dx) {
      int xx = ww + dx; if ((unsigned)xx >= 128u) continue;
      r3 += w3[ch * 9 + (dy + 1) * 3 + (dx + 1)] * base[((size_t)(y * 128 + xx)) * HID2 + ch];
    }
  }
#pragma unroll
  for (int dy = -2; dy <= 2; ++dy) {
    int y = hh + dy; if ((unsigned)y >= 128u) continue;
#pragma unroll
    for (int dx = -2; dx <= 2; ++dx) {
      int xx = ww + dx; if ((unsigned)xx >= 128u) continue;
      r5 += w5[ch * 25 + (dy + 2) * 5 + (dx + 2)] * base[((size_t)(y * 128 + xx)) * HID2 + ch];
    }
  }
  r3 = fmaxf(r3, 0.f); r5 = fmaxf(r5, 0.f);
  if (ch < HIDN) {
    x1o[pos * HID2 + ch] = r3;
    x1o[pos * HID2 + HIDN + ch] = r5;
  } else {
    x2o[pos * HID2 + (ch - HIDN)] = r3;
    x2o[pos * HID2 + ch] = r5;
  }
}

// ---------------------------------------------------------------- FFN grouped dw stage 2 (+GELU)
__global__ void ffn_dw2_kernel(const float* __restrict__ x1, const float* __restrict__ x2,
                               const float* __restrict__ w31, const float* __restrict__ w51,
                               bf16_t* __restrict__ out) {
  size_t idx = (size_t)blockIdx.x * blockDim.x + threadIdx.x;  // pos*768 + g
  if (idx >= (size_t)POSN * HIDN) return;
  int g = (int)(idx % HIDN);
  size_t pos = idx / HIDN;
  int p = (int)(pos & 16383);
  size_t bimg = pos >> 14;
  int hh = p >> 7, ww = p & 127;
  const float* b1 = x1 + (bimg << 14) * HID2;
  const float* b2 = x2 + (bimg << 14) * HID2;
  float a1 = 0.f, a2 = 0.f;
#pragma unroll
  for (int i = 0; i < 2; ++i) {
    int cin = g * 2 + i;
#pragma unroll
    for (int dy = -1; dy <= 1; ++dy) {
      int y = hh + dy; if ((unsigned)y >= 128u) continue;
#pragma unroll
      for (int dx = -1; dx <= 1; ++dx) {
        int xx = ww + dx; if ((unsigned)xx >= 128u) continue;
        a1 += w31[g * 18 + i * 9 + (dy + 1) * 3 + (dx + 1)] * b1[((size_t)(y * 128 + xx)) * HID2 + cin];
      }
    }
#pragma unroll
    for (int dy = -2; dy <= 2; ++dy) {
      int y = hh + dy; if ((unsigned)y >= 128u) continue;
#pragma unroll
      for (int dx = -2; dx <= 2; ++dx) {
        int xx = ww + dx; if ((unsigned)xx >= 128u) continue;
        a2 += w51[g * 50 + i * 25 + (dy + 2) * 5 + (dx + 2)] * b2[((size_t)(y * 128 + xx)) * HID2 + cin];
      }
    }
  }
  out[pos * HID2 + g]        = (bf16_t)gelu_f(a1);
  out[pos * HID2 + HIDN + g] = (bf16_t)gelu_f(a2);
}

// ================================================================ host launch
extern "C" void kernel_launch(void* const* d_in, const int* in_sizes, int n_in,
                              void* d_out, int out_size, void* d_ws, size_t ws_size,
                              hipStream_t stream) {
  const float* x      = (const float*)d_in[0];
  const float* ln1_w  = (const float*)d_in[1];
  const float* ln1_b  = (const float*)d_in[2];
  const float* rpb    = (const float*)d_in[3];
  const float* qkv_w  = (const float*)d_in[4];
  const float* qkv_b  = (const float*)d_in[5];
  const float* qkvdww = (const float*)d_in[6];
  const float* qkvdwb = (const float*)d_in[7];
  const float* q1_w   = (const float*)d_in[8];
  const float* q2_w   = (const float*)d_in[9];
  const float* proj_w = (const float*)d_in[10];
  const float* proj_b = (const float*)d_in[11];
  const float* ln2_w  = (const float*)d_in[12];
  const float* ln2_b  = (const float*)d_in[13];
  const float* pin_w  = (const float*)d_in[14];
  const float* dw3_w  = (const float*)d_in[15];
  const float* dw5_w  = (const float*)d_in[16];
  const float* dw31_w = (const float*)d_in[17];
  const float* dw51_w = (const float*)d_in[18];
  const float* pout_w = (const float*)d_in[19];
  float* out = (float*)d_out;

  uint8_t* wsp = (uint8_t*)d_ws;
  auto alloc = [&](size_t bytes) -> void* {
    void* p = (void*)wsp;
    wsp += (bytes + 255) & ~(size_t)255;
    return p;
  };
  bf16_t* wQKV  = (bf16_t*)alloc((size_t)CC * 576 * 2);
  bf16_t* wQ1   = (bf16_t*)alloc((size_t)CC * CC * 2);
  bf16_t* wQ2   = (bf16_t*)alloc((size_t)CC * CC * 2);
  bf16_t* wPROJ = (bf16_t*)alloc((size_t)CC * CC * 2);
  bf16_t* wPIN  = (bf16_t*)alloc((size_t)CC * HID2 * 2);
  bf16_t* wPOUT = (bf16_t*)alloc((size_t)HID2 * CC * 2);
  float*  biasT = (float*)alloc((size_t)NHEAD * 64 * 64 * 4);
  float*  xw_f  = (float*)alloc((size_t)POSN * CC * 4);
  bf16_t* xw_b  = (bf16_t*)alloc((size_t)POSN * CC * 2);
  bf16_t* xfftb = (bf16_t*)alloc((size_t)POSN * CC * 2);
  bf16_t* g1b   = (bf16_t*)alloc((size_t)POSN * CC * 2);
  float*  xfft2 = (float*)alloc((size_t)POSN * CC * 4);
  float*  xifft = (float*)alloc((size_t)POSN * CC * 4);
  float*  qkv_t = (float*)alloc((size_t)POSN * 576 * 4);
  float*  qkv_d = (float*)alloc((size_t)POSN * 576 * 4);
  bf16_t* qb    = (bf16_t*)alloc((size_t)POSN * CC * 2);
  bf16_t* kb    = (bf16_t*)alloc((size_t)POSN * CC * 2);
  bf16_t* vb    = (bf16_t*)alloc((size_t)POSN * CC * 2);
  bf16_t* attnb = (bf16_t*)alloc((size_t)POSN * CC * 2);
  float*  projo = (float*)alloc((size_t)POSN * CC * 4);
  float*  x2buf = (float*)alloc((size_t)POSN * CC * 4);
  bf16_t* ln2b  = (bf16_t*)alloc((size_t)POSN * CC * 2);
  float*  ffn1  = (float*)alloc((size_t)POSN * HID2 * 4);
  float*  x1m   = (float*)alloc((size_t)POSN * HID2 * 4);
  float*  x2m   = (float*)alloc((size_t)POSN * HID2 * 4);
  bf16_t* ffn2b = (bf16_t*)alloc((size_t)POSN * HID2 * 2);
  float*  ffno  = (float*)alloc((size_t)POSN * CC * 4);

  // --- weight conversion to WMMA fragment layout ---
  auto cvt = [&](const float* W, bf16_t* Wt, int N, int K) {
    size_t n = (size_t)N * K;
    convert_frag_kernel<<<(int)((n + 255) / 256), 256, 0, stream>>>(W, Wt, N, K);
  };
  cvt(qkv_w, wQKV, 576, CC);
  cvt(q1_w, wQ1, CC, CC);
  cvt(q2_w, wQ2, CC, CC);
  cvt(proj_w, wPROJ, CC, CC);
  cvt(pin_w, wPIN, HID2, CC);
  cvt(pout_w, wPOUT, CC, HID2);
  bias_table_kernel<<<(NHEAD * 64 * 64 + 255) / 256, 256, 0, stream>>>(rpb, biasT);

  // --- LN1 + window partition ---
  ln1_window_kernel<<<POSN / 4, 128, 0, stream>>>(x, ln1_w, ln1_b, xw_f, xw_b);

  // --- FFT branch ---
  fft_fwd_kernel<<<NWIN * CC / 4, 256, 0, stream>>>(xw_f, xfftb);
  gemm_bf16_kernel<<<dim3(CC / 64, POSN / 128), 256, 0, stream>>>(
      xfftb, wQ1, nullptr, nullptr, g1b, POSN, CC, CC, /*act=*/1);
  gemm_bf16_kernel<<<dim3(CC / 64, POSN / 128), 256, 0, stream>>>(
      g1b, wQ2, nullptr, xfft2, nullptr, POSN, CC, CC, 0);
  fft_inv_kernel<<<NWIN * CC / 4, 256, 0, stream>>>(xfft2, xifft);

  // --- qkv projection + depthwise + assemble ---
  gemm_bf16_kernel<<<dim3(576 / 64, POSN / 128), 256, 0, stream>>>(
      xw_b, wQKV, qkv_b, qkv_t, nullptr, POSN, 576, CC, 0);
  qkv_dw_kernel<<<(int)(((size_t)POSN * 576 + 255) / 256), 256, 0, stream>>>(
      qkv_t, qkvdww, qkvdwb, qkv_d);
  qkv_assemble_kernel<<<(int)(((size_t)POSN * CC + 255) / 256), 256, 0, stream>>>(
      qkv_d, xifft, qb, kb, vb);

  // --- attention (WMMA) ---
  attention_kernel<<<NWIN * NHEAD, 32, 0, stream>>>(qb, kb, vb, biasT, attnb);

  // --- output projection + window-reverse residual ---
  gemm_bf16_kernel<<<dim3(CC / 64, POSN / 128), 256, 0, stream>>>(
      attnb, wPROJ, proj_b, projo, nullptr, POSN, CC, CC, 0);
  residual1_kernel<<<(int)(((size_t)POSN * CC + 255) / 256), 256, 0, stream>>>(x, projo, x2buf);

  // --- LN2 + FFN ---
  ln2_kernel<<<POSN / 4, 128, 0, stream>>>(x2buf, ln2_w, ln2_b, ln2b);
  gemm_bf16_kernel<<<dim3(HID2 / 64, POSN / 128), 256, 0, stream>>>(
      ln2b, wPIN, nullptr, ffn1, nullptr, POSN, HID2, CC, 0);
  ffn_dw1_kernel<<<(int)(((size_t)POSN * HID2 + 255) / 256), 256, 0, stream>>>(
      ffn1, dw3_w, dw5_w, x1m, x2m);
  ffn_dw2_kernel<<<(int)(((size_t)POSN * HIDN + 255) / 256), 256, 0, stream>>>(
      x1m, x2m, dw31_w, dw51_w, ffn2b);
  gemm_bf16_kernel<<<dim3(CC / 64, POSN / 128), 256, 0, stream>>>(
      ffn2b, wPOUT, nullptr, ffno, nullptr, POSN, CC, HID2, 0);
  residual2_kernel<<<(int)(((size_t)POSN * CC + 255) / 256), 256, 0, stream>>>(x2buf, ffno, out);
}
